// Attention_29626684408460
// MI455X (gfx1250) — matrix operand order repaired and checked by reading, not compiled, not run on hardware
//
#include <hip/hip_runtime.h>

typedef __attribute__((ext_vector_type(16))) __bf16 v16bf;
typedef __attribute__((ext_vector_type(8)))  float  v8f;

#define B_  32
#define S_  2048
#define D_  1024
#define NCHUNK (D_ / 32)                  // 32 k-chunks of 32
#define TILE_USHORTS (NCHUNK * 32 * 16)   // 16384 ushorts = 32KB per 16-row tile

__device__ __forceinline__ unsigned short f32_to_bf16(float f) {
    unsigned int u = __builtin_bit_cast(unsigned int, f);
    u += 0x7FFFu + ((u >> 16) & 1u);      // round-to-nearest-even
    return (unsigned short)(u >> 16);
}

__device__ __forceinline__ float tanh_hw(float x) {
    float r;
    // native CDNA5 transcendental; v_nop covers the TRANS result hazard
    asm("v_tanh_f32 %0, %1\n\tv_nop\n\tv_nop" : "=v"(r) : "v"(x));
    return r;
}

union ABFrag {
    uint4           q[2];
    unsigned short  u[16];
    v16bf           v;
};

// ---------------------------------------------------------------- Wk f32 -> bf16 (row-major copy)
__global__ __launch_bounds__(256) void wkcvt_kernel(const float* __restrict__ Wk,
                                                    unsigned short* __restrict__ WkBf) {
    int i = (blockIdx.x * 256 + threadIdx.x) * 4;   // 1M elements / 4
    float4 f = *(const float4*)(Wk + i);
    ushort4 o;
    o.x = f32_to_bf16(f.x); o.y = f32_to_bf16(f.y);
    o.z = f32_to_bf16(f.z); o.w = f32_to_bf16(f.w);
    *(ushort4*)(WkBf + i) = o;
}

// ---------------------------------------------------------------- pq = query @ Wq^T
__global__ __launch_bounds__(256) void pq_kernel(const float* __restrict__ query,
                                                 const float* __restrict__ Wq,
                                                 float* __restrict__ pq) {
    int idx = blockIdx.x * 256 + threadIdx.x;       // 0 .. 32767
    int b = idx >> 10, e = idx & (D_ - 1);
    const float4* q = (const float4*)(query + (size_t)b * D_);
    const float4* w = (const float4*)(Wq + (size_t)e * D_);
    float s = 0.f;
    for (int i = 0; i < D_ / 4; ++i) {
        float4 a = q[i], c = w[i];
        s += a.x * c.x + a.y * c.y + a.z * c.z + a.w * c.w;
    }
    pq[idx] = s;
}

// ---------------------------------------------------------------- scores[b,s] (pre-softmax)
// One WG = 32 S-rows (two 16-row A tiles), 8 waves.
// Each wave owns 8 contiguous e-tiles, processed as 2 passes of 4 concurrent
// e-tiles so each A-fragment LDS read feeds 8 WMMAs (0.5 ds_load_b128/WMMA).
// B fragments of a chunk are batch-loaded before the WMMA burst so the wave
// never has to drain loadcnt to zero on the critical path.
__global__ __launch_bounds__(256) void scores_kernel(const float* __restrict__ keys,
                                                     const unsigned short* __restrict__ WkBf,
                                                     const float* __restrict__ pqAll,
                                                     const float* __restrict__ We,
                                                     float* __restrict__ scores) {
    // keys tiles staged in WMMA-fragment order: [tile][chunk][lane][16 bf16]
    __shared__ __align__(16) unsigned short aLds[2 * TILE_USHORTS];   // 64 KB

    const int b   = blockIdx.y;
    const int s0  = blockIdx.x * 32;
    const int tid = threadIdx.x;

    // ---- stage 32 x 1024 keys -> bf16, swizzled to per-lane fragment layout
    const float* kBase = keys + ((size_t)b * S_ + s0) * D_;
    for (int i = tid; i < 32 * D_; i += 256) {
        int rg = i >> 10;            // global row in WG tile (0..31)
        int k  = i & (D_ - 1);       // column
        int tile = rg >> 4, r = rg & 15;
        int c = k >> 5, j = k & 31;
        int lhalf = (j >> 4) & 1;
        int lane  = r + (lhalf << 4);
        int idx   = j & 15;                      // contiguous k = 16*half + idx (matches B)
        aLds[tile * TILE_USHORTS + (c * 32 + lane) * 16 + idx] =
            f32_to_bf16(kBase[(size_t)rg * D_ + k]);
    }
    __syncthreads();

    const int lane  = tid & 31;
    const int wave  = tid >> 5;
    const int lhalf = lane >> 4;     // 0: lanes 0-15, 1: lanes 16-31
    const int lmod  = lane & 15;

    float sAcc0[8], sAcc1[8];
#pragma unroll
    for (int v = 0; v < 8; ++v) { sAcc0[v] = 0.f; sAcc1[v] = 0.f; }

    const uint4* a0Base = (const uint4*)(aLds + (size_t)lane * 16);
    const uint4* a1Base = (const uint4*)(aLds + TILE_USHORTS + (size_t)lane * 16);

    for (int pass = 0; pass < 2; ++pass) {
        v8f acc0[4] = {};            // [e-group][row 0..7]   rows 0..15
        v8f acc1[4] = {};            // [e-group][row 0..7]   rows 16..31

        // 4 concurrent e-tiles: et = wave*8 + pass*4 + g
        int bOff[4];
#pragma unroll
        for (int g = 0; g < 4; ++g) {
            int e0 = (wave * 8 + pass * 4 + g) << 4;
            bOff[g] = (e0 + lmod) * D_ + lhalf * 16;   // bf16 element offset
        }

#pragma unroll 2
        for (int c = 0; c < NCHUNK; ++c) {
            // ---- batch-issue all loads for this chunk first
            ABFrag a0, a1, bf[4];
            a0.q[0] = a0Base[c * 64]; a0.q[1] = a0Base[c * 64 + 1];
            a1.q[0] = a1Base[c * 64]; a1.q[1] = a1Base[c * 64 + 1];
#pragma unroll
            for (int g = 0; g < 4; ++g) {
                const uint4* bp = (const uint4*)(WkBf + bOff[g] + c * 32);
                bf[g].q[0] = bp[0];
                bf[g].q[1] = bp[1];
            }

            // ---- WMMA burst: 8 matrix ops per chunk
#pragma unroll
            for (int g = 0; g < 4; ++g) {
                acc0[g] = __builtin_amdgcn_wmma_f32_16x16x32_bf16(
                    false, a0.v, false, bf[g].v, (short)0, acc0[g], false, false);
                acc1[g] = __builtin_amdgcn_wmma_f32_16x16x32_bf16(
                    false, a1.v, false, bf[g].v, (short)0, acc1[g], false, false);
            }
        }

        // epilogue: acc[g][v] = pk[row = v + 8*lhalf, col e]
#pragma unroll
        for (int g = 0; g < 4; ++g) {
            const int   e = ((wave * 8 + pass * 4 + g) << 4) + lmod;
            const float w = We[e];
            const float p = pqAll[b * D_ + e];
#pragma unroll
            for (int v = 0; v < 8; ++v) {
                sAcc0[v] += w * tanh_hw(p + acc0[g][v]);
                sAcc1[v] += w * tanh_hw(p + acc1[g][v]);
            }
        }
    }

    // ---- reduce 16 e-columns within each half-wave group
#pragma unroll
    for (int m = 1; m < 16; m <<= 1) {
#pragma unroll
        for (int v = 0; v < 8; ++v) {
            sAcc0[v] += __shfl_xor(sAcc0[v], m, 32);
            sAcc1[v] += __shfl_xor(sAcc1[v], m, 32);
        }
    }

    // ---- combine across waves: reuse aLds as a 32-float accumulator (ds_add_f32)
    __syncthreads();                       // everyone done reading aLds
    float* sRed = (float*)aLds;
    if (tid < 32) sRed[tid] = 0.f;
    __syncthreads();
    if (lmod == 0) {
#pragma unroll
        for (int v = 0; v < 8; ++v) {
            atomicAdd(&sRed[v + 8 * lhalf],      sAcc0[v]);   // rows 0..15
            atomicAdd(&sRed[16 + v + 8 * lhalf], sAcc1[v]);   // rows 16..31
        }
    }
    __syncthreads();
    if (tid < 32)
        scores[(size_t)b * S_ + s0 + tid] = sRed[tid];
}

// ---------------------------------------------------------------- softmax over S per batch
__global__ __launch_bounds__(256) void softmax_kernel(const float* __restrict__ scores,
                                                      float* __restrict__ out) {
    __shared__ float red[256];
    const int b   = blockIdx.x;
    const int tid = threadIdx.x;
    const float* row = scores + (size_t)b * S_;

    float v[8];
    float mx = -__builtin_inff();
#pragma unroll
    for (int i = 0; i < 8; ++i) {
        v[i] = row[tid + i * 256];
        mx = fmaxf(mx, v[i]);
    }
    red[tid] = mx;
    __syncthreads();
    for (int s = 128; s > 0; s >>= 1) {
        if (tid < s) red[tid] = fmaxf(red[tid], red[tid + s]);
        __syncthreads();
    }
    mx = red[0];
    __syncthreads();

    float sum = 0.f;
#pragma unroll
    for (int i = 0; i < 8; ++i) {
        v[i] = __expf(v[i] - mx);
        sum += v[i];
    }
    red[tid] = sum;
    __syncthreads();
    for (int s = 128; s > 0; s >>= 1) {
        if (tid < s) red[tid] += red[tid + s];
        __syncthreads();
    }
    const float inv = 1.f / red[0];
#pragma unroll
    for (int i = 0; i < 8; ++i)
        out[(size_t)b * S_ + tid + i * 256] = v[i] * inv;
}

// ---------------------------------------------------------------- launch
extern "C" void kernel_launch(void* const* d_in, const int* in_sizes, int n_in,
                              void* d_out, int out_size, void* d_ws, size_t ws_size,
                              hipStream_t stream) {
    const float* query = (const float*)d_in[0];   // [B, QD]
    const float* keys  = (const float*)d_in[1];   // [B, S, KD]
    const float* Wq    = (const float*)d_in[2];   // [QD, QD]
    const float* Wk    = (const float*)d_in[3];   // [QD, KD]
    const float* We    = (const float*)d_in[4];   // [1, QD]
    float* out = (float*)d_out;                   // [B, S]

    float*          pq     = (float*)d_ws;                         // B*D f32 (128 KB)
    float*          scores = pq + B_ * D_;                         // B*S f32 (256 KB)
    unsigned short* WkBf   = (unsigned short*)(scores + B_ * S_);  // D*D bf16 (2 MB)

    wkcvt_kernel<<<(D_ * D_) / (256 * 4), 256, 0, stream>>>(Wk, WkBf);
    pq_kernel<<<(B_ * D_) / 256, 256, 0, stream>>>(query, Wq, pq);

    dim3 grid(S_ / 32, B_);
    scores_kernel<<<grid, 256, 0, stream>>>(keys, WkBf, pq, We, scores);

    softmax_kernel<<<B_, 256, 0, stream>>>(scores, out);
}
